// SGDAT_56435870269991
// MI455X (gfx1250) — matrix operand order repaired and checked by reading, not compiled, not run on hardware
//
#include <hip/hip_runtime.h>
#include <hip/hip_bf16.h>

typedef __attribute__((ext_vector_type(16))) __bf16 v16bf;
typedef __attribute__((ext_vector_type(8)))  __bf16 v8bf;
typedef __attribute__((ext_vector_type(8)))  float  v8f;

#define B_   32
#define N_   8192
#define C_   64
#define S_   512
#define CIN  67
#define KP1  96     // cin padded to multiple of 32
#define CO   128
#define HID  8
#define BN_EPS 1e-3f

// swizzled-B sizes: (K/32) k-tiles x 8 n-tiles x 32 lanes x 16 bf16
#define SWB1_ELEMS (3 * 8 * 32 * 16)   // 12288 (K=96)
#define SWB2_ELEMS (4 * 8 * 32 * 16)   // 16384 (K=128)

// ---------------------------------------------------------------------------
// WMMA helpers (CDNA5 v_wmma_f32_16x16x32_bf16, wave32)
// ---------------------------------------------------------------------------
__device__ __forceinline__ v8f wmma_bf16(v16bf a, v16bf b, v8f c) {
  return __builtin_amdgcn_wmma_f32_16x16x32_bf16(
      /*neg_a=*/false, a, /*neg_b=*/false, b,
      /*c_mod=*/(short)0, c, /*reuse_a=*/false, /*reuse_b=*/false);
}

// A fragment: 16x32 bf16, row-major LDS source with leading dim ld (mult of 8).
// lane m=lane&15; lanes<16 hold K {0..15}, lanes>=16 hold {16..31}.
// Two aligned 16-byte LDS loads per lane.
__device__ __forceinline__ v16bf fragA(const __bf16* a, int ld, int k0, int lane) {
  int m  = lane & 15;
  int kb = (lane >> 4) * 16;
  const v8bf* p = (const v8bf*)(a + m * ld + k0 + kb);
  v8bf lo = p[0];
  v8bf hi = p[1];   // +8 elements
  return __builtin_shufflevector(lo, hi, 0, 1, 2, 3, 4, 5, 6, 7,
                                 8, 9, 10, 11, 12, 13, 14, 15);
}

// B fragment from pre-swizzled storage: ((kt*8 + nt)*32 + lane)*16 contiguous.
__device__ __forceinline__ v16bf fragBsw(const __bf16* sB, int kt, int nt, int lane) {
  const v8bf* p = (const v8bf*)(sB + (((kt * 8 + nt) * 32 + lane) << 4));
  v8bf lo = p[0];
  v8bf hi = p[1];
  return __builtin_shufflevector(lo, hi, 0, 1, 2, 3, 4, 5, 6, 7,
                                 8, 9, 10, 11, 12, 13, 14, 15);
}

// ---------------------------------------------------------------------------
// Kernel 1: farthest point sampling, one workgroup per batch, points in VGPRs
// ---------------------------------------------------------------------------
__global__ __launch_bounds__(1024) void k_fps(const float* __restrict__ xyz,
                                              int* __restrict__ idxOut) {
  const int b = blockIdx.x;
  const int t = threadIdx.x;                       // 1024 threads, 8 pts each
  const float* base = xyz + (size_t)b * N_ * 3;

  float px[8], py[8], pz[8], dist[8];
#pragma unroll
  for (int j = 0; j < 8; ++j) {
    int p = j * 1024 + t;
    px[j] = base[p * 3 + 0];
    py[j] = base[p * 3 + 1];
    pz[j] = base[p * 3 + 2];
    dist[j] = 1e10f;
  }

  __shared__ float cx, cy, cz;
  __shared__ int   sFar;
  __shared__ float rv[32], rx[32], ry[32], rz[32];
  __shared__ int   ri[32];

  if (t == 0) { sFar = 0; cx = base[0]; cy = base[1]; cz = base[2]; }
  __syncthreads();

  const int wid = t >> 5, lane = t & 31;

  for (int it = 0; it < S_; ++it) {
    if (t == 0) idxOut[b * S_ + it] = sFar;
    const float ccx = cx, ccy = cy, ccz = cz;

    float bv = -1.0f, bx = 0.f, by = 0.f, bz = 0.f;
    int   bi = 0x7fffffff;
#pragma unroll
    for (int j = 0; j < 8; ++j) {
      float dx = px[j] - ccx, dy = py[j] - ccy, dz = pz[j] - ccz;
      float d  = dx * dx + dy * dy + dz * dz;
      dist[j]  = fminf(dist[j], d);
      int p = j * 1024 + t;
      if (dist[j] > bv || (dist[j] == bv && p < bi)) {
        bv = dist[j]; bi = p; bx = px[j]; by = py[j]; bz = pz[j];
      }
    }
    // wave32 shuffle argmax (carry coords so no re-gather needed)
#pragma unroll
    for (int off = 16; off > 0; off >>= 1) {
      float v2 = __shfl_down(bv, off);
      int   i2 = __shfl_down(bi, off);
      float x2 = __shfl_down(bx, off);
      float y2 = __shfl_down(by, off);
      float z2 = __shfl_down(bz, off);
      if (v2 > bv || (v2 == bv && i2 < bi)) { bv = v2; bi = i2; bx = x2; by = y2; bz = z2; }
    }
    if (lane == 0) { rv[wid] = bv; ri[wid] = bi; rx[wid] = bx; ry[wid] = by; rz[wid] = bz; }
    __syncthreads();
    if (wid == 0) {
      bv = rv[lane]; bi = ri[lane]; bx = rx[lane]; by = ry[lane]; bz = rz[lane];
#pragma unroll
      for (int off = 16; off > 0; off >>= 1) {
        float v2 = __shfl_down(bv, off);
        int   i2 = __shfl_down(bi, off);
        float x2 = __shfl_down(bx, off);
        float y2 = __shfl_down(by, off);
        float z2 = __shfl_down(bz, off);
        if (v2 > bv || (v2 == bv && i2 < bi)) { bv = v2; bi = i2; bx = x2; by = y2; bz = z2; }
      }
      if (lane == 0) { sFar = bi; cx = bx; cy = by; cz = bz; }
    }
    __syncthreads();
  }
}

// ---------------------------------------------------------------------------
// Kernel 2: pack weights to bf16 *fragment-swizzled* B layout + fold BN
// layout index: ((kt*8 + nt)*32 + lane)*16 + e
//   n = nt*16 + (lane&15) ; k = kt*32 + (lane>>4)*16 + e
// ---------------------------------------------------------------------------
__global__ void k_pack(const float* __restrict__ W1, const float* __restrict__ W2,
                       const float* __restrict__ Wp,
                       const float* __restrict__ g1, const float* __restrict__ b1,
                       const float* __restrict__ m1, const float* __restrict__ v1,
                       const float* __restrict__ g2, const float* __restrict__ b2,
                       const float* __restrict__ m2, const float* __restrict__ v2,
                       __bf16* __restrict__ wB1, __bf16* __restrict__ wB2,
                       __bf16* __restrict__ wBp,
                       float* __restrict__ sc1, float* __restrict__ bi1,
                       float* __restrict__ sc2, float* __restrict__ bi2) {
  for (int i = threadIdx.x; i < SWB1_ELEMS; i += blockDim.x) {
    int e  = i & 15;
    int l  = (i >> 4) & 31;
    int nt = (i >> 9) & 7;
    int kt = i >> 12;
    int n  = nt * 16 + (l & 15);
    int k  = kt * 32 + (l >> 4) * 16 + e;
    wB1[i] = (k < CIN) ? (__bf16)W1[n * CIN + k] : (__bf16)0.0f;
    wBp[i] = (k < CIN) ? (__bf16)Wp[n * CIN + k] : (__bf16)0.0f;
  }
  for (int i = threadIdx.x; i < SWB2_ELEMS; i += blockDim.x) {
    int e  = i & 15;
    int l  = (i >> 4) & 31;
    int nt = (i >> 9) & 7;
    int kt = i >> 12;
    int n  = nt * 16 + (l & 15);
    int k  = kt * 32 + (l >> 4) * 16 + e;
    wB2[i] = (__bf16)W2[n * CO + k];
  }
  for (int c = threadIdx.x; c < CO; c += blockDim.x) {
    float s1 = g1[c] * rsqrtf(v1[c] + BN_EPS);
    sc1[c] = s1; bi1[c] = b1[c] - m1[c] * s1;
    float s2 = g2[c] * rsqrtf(v2[c] + BN_EPS);
    sc2[c] = s2; bi2[c] = b2[c] - m2[c] * s2;
  }
}

// ---------------------------------------------------------------------------
// Kernel 3: gather sampled points, concat [xyz|feat], zero-pad to K=96, bf16
// ---------------------------------------------------------------------------
__global__ void k_gather(const float* __restrict__ xyz, const float* __restrict__ feat,
                         const int* __restrict__ idx, __bf16* __restrict__ xp) {
  int i = blockIdx.x * blockDim.x + threadIdx.x;   // over B*S*KP1
  if (i >= B_ * S_ * KP1) return;
  int c  = i % KP1;
  int bs = i / KP1;
  int s  = bs % S_;
  int b  = bs / S_;
  int p  = idx[b * S_ + s];
  float v = 0.0f;
  if (c < 3)        v = xyz [((size_t)b * N_ + p) * 3  + c];
  else if (c < CIN) v = feat[((size_t)b * N_ + p) * C_ + (c - 3)];
  xp[i] = (__bf16)v;
}

// ---------------------------------------------------------------------------
// Kernel 4: h1 = relu(bn1(x @ W1^T))   (16384x96)x(96x128), WMMA bf16
// block = 256 thr (8 waves) -> 64x128 tile; wave -> 16x64 (4 accum tiles)
// ---------------------------------------------------------------------------
__global__ __launch_bounds__(256) void k_gemm1(const __bf16* __restrict__ xp,
                                               const __bf16* __restrict__ wB,
                                               const float* __restrict__ sc,
                                               const float* __restrict__ bi,
                                               __bf16* __restrict__ h1) {
  __shared__ v8bf sAv[64 * KP1 / 8];
  __shared__ v8bf sBv[SWB1_ELEMS / 8];
  const __bf16* sA = (const __bf16*)sAv;
  const __bf16* sB = (const __bf16*)sBv;
  const int row0 = blockIdx.x * 64;
  __builtin_prefetch(wB, 0, 2);
  {
    const v8bf* gA = (const v8bf*)(xp + (size_t)row0 * KP1);
    for (int i = threadIdx.x; i < 64 * KP1 / 8; i += 256) sAv[i] = gA[i];
    const v8bf* gB = (const v8bf*)wB;
    for (int i = threadIdx.x; i < SWB1_ELEMS / 8; i += 256) sBv[i] = gB[i];
  }
  __syncthreads();

  const int w = threadIdx.x >> 5, lane = threadIdx.x & 31;
  const int m0 = (w & 3) * 16, ntb = (w >> 2) * 4;
  v8f acc[4] = {};
#pragma unroll
  for (int kt = 0; kt < 3; ++kt) {
    v16bf a = fragA(sA + m0 * KP1, KP1, kt * 32, lane);
#pragma unroll
    for (int t = 0; t < 4; ++t) {
      v16bf bf = fragBsw(sB, kt, ntb + t, lane);
      acc[t] = wmma_bf16(a, bf, acc[t]);
    }
  }
  const int n = lane & 15, mh = (lane >> 4) * 8;
#pragma unroll
  for (int t = 0; t < 4; ++t) {
    int col = (ntb + t) * 16 + n;
    float s = sc[col], bb = bi[col];
#pragma unroll
    for (int v = 0; v < 8; ++v) {
      int r = row0 + m0 + mh + v;
      float val = fmaxf(acc[t][v] * s + bb, 0.0f);
      h1[(size_t)r * CO + col] = (__bf16)val;
    }
  }
}

// ---------------------------------------------------------------------------
// Kernel 5: h2 = bn2(h1 @ W2^T)   (16384x128)x(128x128), WMMA bf16, f32 out
// ---------------------------------------------------------------------------
__global__ __launch_bounds__(256) void k_gemm2(const __bf16* __restrict__ h1,
                                               const __bf16* __restrict__ wB,
                                               const float* __restrict__ sc,
                                               const float* __restrict__ bi,
                                               float* __restrict__ h2) {
  __shared__ v8bf sAv[64 * CO / 8];
  __shared__ v8bf sBv[SWB2_ELEMS / 8];
  const __bf16* sA = (const __bf16*)sAv;
  const __bf16* sB = (const __bf16*)sBv;
  const int row0 = blockIdx.x * 64;
  __builtin_prefetch(wB, 0, 2);
  {
    const v8bf* gA = (const v8bf*)(h1 + (size_t)row0 * CO);
    for (int i = threadIdx.x; i < 64 * CO / 8; i += 256) sAv[i] = gA[i];
    const v8bf* gB = (const v8bf*)wB;
    for (int i = threadIdx.x; i < SWB2_ELEMS / 8; i += 256) sBv[i] = gB[i];
  }
  __syncthreads();

  const int w = threadIdx.x >> 5, lane = threadIdx.x & 31;
  const int m0 = (w & 3) * 16, ntb = (w >> 2) * 4;
  v8f acc[4] = {};
#pragma unroll
  for (int kt = 0; kt < 4; ++kt) {
    v16bf a = fragA(sA + m0 * CO, CO, kt * 32, lane);
#pragma unroll
    for (int t = 0; t < 4; ++t) {
      v16bf bf = fragBsw(sB, kt, ntb + t, lane);
      acc[t] = wmma_bf16(a, bf, acc[t]);
    }
  }
  const int n = lane & 15, mh = (lane >> 4) * 8;
#pragma unroll
  for (int t = 0; t < 4; ++t) {
    int col = (ntb + t) * 16 + n;
    float s = sc[col], bb = bi[col];
#pragma unroll
    for (int v = 0; v < 8; ++v) {
      int r = row0 + m0 + mh + v;
      h2[(size_t)r * CO + col] = acc[t][v] * s + bb;
    }
  }
}

// ---------------------------------------------------------------------------
// Kernel 6: SE gate — mean over S, 128->8->128 MLP, sigmoid. 1 block / batch
// ---------------------------------------------------------------------------
__global__ __launch_bounds__(128) void k_se(const float* __restrict__ h2,
                                            const float* __restrict__ Wse1,
                                            const float* __restrict__ bse1,
                                            const float* __restrict__ Wse2,
                                            const float* __restrict__ bse2,
                                            float* __restrict__ gates) {
  const int b = blockIdx.x, c = threadIdx.x;       // 128 threads
  const float* p = h2 + (size_t)b * S_ * CO + c;
  float sum = 0.0f;
  for (int s = 0; s < S_; ++s) sum += p[s * CO];
  __shared__ float pool[CO];
  __shared__ float hid[HID];
  pool[c] = sum * (1.0f / (float)S_);
  __syncthreads();
  if (c < HID) {
    float a = bse1[c];
    for (int k = 0; k < CO; ++k) a += Wse1[c * CO + k] * pool[k];
    hid[c] = fmaxf(a, 0.0f);
  }
  __syncthreads();
  float a = bse2[c];
#pragma unroll
  for (int k = 0; k < HID; ++k) a += Wse2[c * HID + k] * hid[k];
  gates[b * CO + c] = 1.0f / (1.0f + __expf(-a));
}

// ---------------------------------------------------------------------------
// Kernel 7: out = relu(h2 * gate + x @ Wproj^T)   proj GEMM fused with gate
// ---------------------------------------------------------------------------
__global__ __launch_bounds__(256) void k_final(const __bf16* __restrict__ xp,
                                               const __bf16* __restrict__ wBp,
                                               const float* __restrict__ h2,
                                               const float* __restrict__ gates,
                                               float* __restrict__ out) {
  __shared__ v8bf sAv[64 * KP1 / 8];
  __shared__ v8bf sBv[SWB1_ELEMS / 8];
  const __bf16* sA = (const __bf16*)sAv;
  const __bf16* sB = (const __bf16*)sBv;
  const int row0 = blockIdx.x * 64;
  const int b = row0 / S_;                         // 64 rows never span batches
  __builtin_prefetch(wBp, 0, 2);
  {
    const v8bf* gA = (const v8bf*)(xp + (size_t)row0 * KP1);
    for (int i = threadIdx.x; i < 64 * KP1 / 8; i += 256) sAv[i] = gA[i];
    const v8bf* gB = (const v8bf*)wBp;
    for (int i = threadIdx.x; i < SWB1_ELEMS / 8; i += 256) sBv[i] = gB[i];
  }
  __syncthreads();

  const int w = threadIdx.x >> 5, lane = threadIdx.x & 31;
  const int m0 = (w & 3) * 16, ntb = (w >> 2) * 4;
  v8f acc[4] = {};
#pragma unroll
  for (int kt = 0; kt < 3; ++kt) {
    v16bf a = fragA(sA + m0 * KP1, KP1, kt * 32, lane);
#pragma unroll
    for (int t = 0; t < 4; ++t) {
      v16bf bf = fragBsw(sB, kt, ntb + t, lane);
      acc[t] = wmma_bf16(a, bf, acc[t]);
    }
  }
  const int n = lane & 15, mh = (lane >> 4) * 8;
#pragma unroll
  for (int t = 0; t < 4; ++t) {
    int col = (ntb + t) * 16 + n;
    float g = gates[b * CO + col];
#pragma unroll
    for (int v = 0; v < 8; ++v) {
      int r = row0 + m0 + mh + v;
      float val = h2[(size_t)r * CO + col] * g + acc[t][v];
      out[(size_t)r * CO + col] = fmaxf(val, 0.0f);
    }
  }
}

// ---------------------------------------------------------------------------
// Host launcher
// ---------------------------------------------------------------------------
static inline size_t alignUp(size_t x, size_t a) { return (x + a - 1) & ~(a - 1); }

extern "C" void kernel_launch(void* const* d_in, const int* in_sizes, int n_in,
                              void* d_out, int out_size, void* d_ws, size_t ws_size,
                              hipStream_t stream) {
  (void)in_sizes; (void)n_in; (void)out_size; (void)ws_size;
  const float* xyz  = (const float*)d_in[0];
  const float* feat = (const float*)d_in[1];
  const float* W1   = (const float*)d_in[2];
  const float* g1   = (const float*)d_in[3];
  const float* b1   = (const float*)d_in[4];
  const float* m1   = (const float*)d_in[5];
  const float* v1   = (const float*)d_in[6];
  const float* W2   = (const float*)d_in[7];
  const float* g2   = (const float*)d_in[8];
  const float* b2   = (const float*)d_in[9];
  const float* m2   = (const float*)d_in[10];
  const float* v2   = (const float*)d_in[11];
  const float* Wp   = (const float*)d_in[12];
  const float* Wse1 = (const float*)d_in[13];
  const float* bse1 = (const float*)d_in[14];
  const float* Wse2 = (const float*)d_in[15];
  const float* bse2 = (const float*)d_in[16];
  float* out = (float*)d_out;

  // workspace carve-out (aligned to 256B)
  char* ws = (char*)d_ws;
  size_t off = 0;
  int*    idx  = (int*)(ws + off);    off = alignUp(off + (size_t)B_ * S_ * 4, 256);
  __bf16* xp   = (__bf16*)(ws + off); off = alignUp(off + (size_t)B_ * S_ * KP1 * 2, 256);
  __bf16* h1   = (__bf16*)(ws + off); off = alignUp(off + (size_t)B_ * S_ * CO * 2, 256);
  float*  h2   = (float*)(ws + off);  off = alignUp(off + (size_t)B_ * S_ * CO * 4, 256);
  float*  gts  = (float*)(ws + off);  off = alignUp(off + (size_t)B_ * CO * 4, 256);
  __bf16* wB1  = (__bf16*)(ws + off); off = alignUp(off + (size_t)SWB1_ELEMS * 2, 256);
  __bf16* wB2  = (__bf16*)(ws + off); off = alignUp(off + (size_t)SWB2_ELEMS * 2, 256);
  __bf16* wBp  = (__bf16*)(ws + off); off = alignUp(off + (size_t)SWB1_ELEMS * 2, 256);
  float*  sc1  = (float*)(ws + off);  off = alignUp(off + CO * 4, 256);
  float*  bi1  = (float*)(ws + off);  off = alignUp(off + CO * 4, 256);
  float*  sc2  = (float*)(ws + off);  off = alignUp(off + CO * 4, 256);
  float*  bi2  = (float*)(ws + off);  off = alignUp(off + CO * 4, 256);

  k_fps<<<B_, 1024, 0, stream>>>(xyz, idx);
  k_pack<<<1, 256, 0, stream>>>(W1, W2, Wp, g1, b1, m1, v1, g2, b2, m2, v2,
                                wB1, wB2, wBp, sc1, bi1, sc2, bi2);
  k_gather<<<(B_ * S_ * KP1 + 255) / 256, 256, 0, stream>>>(xyz, feat, idx, xp);
  k_gemm1<<<B_ * S_ / 64, 256, 0, stream>>>(xp, wB1, sc1, bi1, h1);
  k_gemm2<<<B_ * S_ / 64, 256, 0, stream>>>(h1, wB2, sc2, bi2, h2);
  k_se<<<B_, 128, 0, stream>>>(h2, Wse1, bse1, Wse2, bse2, gts);
  k_final<<<B_ * S_ / 64, 256, 0, stream>>>(xp, wBp, h2, gts, out);
}
